// CS_KNN_3D_58557584113736
// MI455X (gfx1250) — matrix operand order repaired and checked by reading, not compiled, not run on hardware
//
#include <hip/hip_runtime.h>
#include <hip/hip_bf16.h>

typedef __attribute__((ext_vector_type(16))) _Float16 v16h;
typedef __attribute__((ext_vector_type(8)))  _Float16 v8h;
typedef __attribute__((ext_vector_type(8)))  float    v8f;

#define EMBED 384
#define NE    128
#define TOPKK 64
#define NN    16384
#define NB    4

union V16H { v16h v; v8h h[2]; };

// --------------------------------------------------------------------------
// Tensor Data Mover support (gfx1250). This toolchain (clang-23 / therock
// HIP headers) exposes the 6-arg builtin:
//   (uint32x4 g0, int32x8 g1, int32x4, int32x4, int32x8, i32 cpol)
// --------------------------------------------------------------------------
#if defined(__has_builtin)
#if __has_builtin(__builtin_amdgcn_tensor_load_to_lds) && \
    __has_builtin(__builtin_amdgcn_s_wait_tensorcnt)
#define USE_TDM 1
#endif
#endif
#ifndef USE_TDM
#define USE_TDM 0
#endif

#if USE_TDM
typedef unsigned int v4u __attribute__((ext_vector_type(4)));
typedef int          v8i_t __attribute__((ext_vector_type(8)));
typedef int          v4i_t __attribute__((ext_vector_type(4)));

// DMA a 2D tile: 128 rows x 32 halves (64B), row stride EMBED halves,
// from global gsrc into LDS at ldst (contiguous [row][k] layout).
__device__ __forceinline__ void tdm_load_b(const _Float16* gsrc,
                                           const _Float16* ldst) {
    unsigned long long ga = (unsigned long long)gsrc;
    typedef __attribute__((address_space(3))) void lds_t;
    unsigned int la = (unsigned int)(unsigned long long)(lds_t*)ldst;

    // D# group0: [1:0]=count=1 | lds_addr[63:32] | global_addr[120:64] | type=2
    v4u g0 = { 1u,
               la,
               (unsigned int)(ga & 0xFFFFFFFFull),
               (unsigned int)((ga >> 32) & 0x01FFFFFFull) | 0x80000000u };
    // D# group1:
    //  w0: wg_mask=0, data_size=1 (2 bytes) in [17:16]
    //  w1: tensor_dim0 lo16 in [31:16]  (=384)
    //  w2: tensor_dim0 hi=0, tensor_dim1 lo16=0
    //  w3: tensor_dim1 hi16=16 (=> 1M rows, no clipping), tile_dim0=32 in [31:16]
    //  w4: tile_dim1=128 rows, tile_dim2=0
    //  w5: tensor_dim0_stride lo32 = 384 (halves)
    //  w6,w7: stride hi bits / dim1 stride = 0
    v8i_t g1 = { (int)(1u << 16),
                 (int)(384u << 16),
                 0,
                 (int)(16u | (32u << 16)),
                 128,
                 384,
                 0, 0 };
    v4i_t zz = { 0, 0, 0, 0 };
    v8i_t z8 = { 0, 0, 0, 0, 0, 0, 0, 0 };
    __builtin_amdgcn_tensor_load_to_lds(g0, g1, zz, zz, z8, 0);
}
#endif

// ---------------------------------------------------------------------------
// Kernel 0: zero the output (harness poisons d_out with 0xAA)
// ---------------------------------------------------------------------------
__global__ void k_zero(float* __restrict__ out, long n4) {
    long i = (long)blockIdx.x * blockDim.x + threadIdx.x;
    if (i < n4) {
        float4* o = (float4*)out;
        o[i] = make_float4(0.f, 0.f, 0.f, 0.f);
    }
}

// ---------------------------------------------------------------------------
// Kernel 1: per-node L2 normalize (emit f16 copy) + class-token scores
// ---------------------------------------------------------------------------
__global__ void __launch_bounds__(256)
k_normalize(const float* __restrict__ nodes,
            const float* __restrict__ ct,
            const float* __restrict__ temp,
            _Float16* __restrict__ nodes_h,
            float* __restrict__ scores)
{
    int w    = threadIdx.x >> 5;
    int lane = threadIdx.x & 31;
    long idx = (long)blockIdx.x * 8 + w;          // 0 .. B*N-1
    const float* src = nodes + idx * EMBED;

    float x[12], c[12];
    float ss = 0.f, sd = 0.f, sc = 0.f;
#pragma unroll
    for (int j = 0; j < 12; ++j) {
        x[j] = src[lane + j * 32];
        c[j] = ct[lane + j * 32];
        ss += x[j] * x[j];
        sd += x[j] * c[j];
        sc += c[j] * c[j];
    }
#pragma unroll
    for (int m = 16; m >= 1; m >>= 1) {
        ss += __shfl_xor(ss, m, 32);
        sd += __shfl_xor(sd, m, 32);
        sc += __shfl_xor(sc, m, 32);
    }
    float nn   = fmaxf(sqrtf(ss), 1e-12f);
    float nc   = fmaxf(sqrtf(sc), 1e-12f);
    float invn = 1.f / nn;

    _Float16* dst = nodes_h + idx * EMBED;
#pragma unroll
    for (int j = 0; j < 12; ++j) dst[lane + j * 32] = (_Float16)(x[j] * invn);
    if (lane == 0) scores[idx] = sd / (nn * nc * temp[0]);
}

// ---------------------------------------------------------------------------
// Kernel 2: top-128 centers per batch + per-center coords / 1/maxdist
// ---------------------------------------------------------------------------
__global__ void __launch_bounds__(256)
k_topk_centers(const float* __restrict__ scores,
               int* __restrict__ cidx,
               float4* __restrict__ cmeta)
{
    __shared__ float smax[256];
    __shared__ int   sidx[256];
    int t = threadIdx.x;
    int b = blockIdx.x;
    const float* row = scores + (long)b * NN;

    float vals[64];
#pragma unroll
    for (int j = 0; j < 64; ++j) vals[j] = row[j * 256 + t];

    for (int r = 0; r < NE; ++r) {
        float best = -1e30f; int bj = 0;
#pragma unroll
        for (int j = 0; j < 64; ++j) if (vals[j] > best) { best = vals[j]; bj = j; }
        smax[t] = best; sidx[t] = bj * 256 + t;
        __syncthreads();
        for (int s = 128; s >= 1; s >>= 1) {
            if (t < s) {
                float om = smax[t + s]; int oi = sidx[t + s];
                if (om > smax[t] || (om == smax[t] && oi < sidx[t])) {
                    smax[t] = om; sidx[t] = oi;
                }
            }
            __syncthreads();
        }
        int nbest = sidx[0];
        if ((nbest & 255) == t) {
            int jj = nbest >> 8;
#pragma unroll
            for (int j = 0; j < 64; ++j) if (j == jj) vals[j] = -1e30f;
        }
        if (t == 0) {
            cidx[b * NE + r] = nbest;
            float cz = 2.f * (float)(nbest >> 10);
            float cy = (float)((nbest >> 5) & 31);
            float cx = (float)(nbest & 31);
            float mz = fmaxf(cz, 30.f - cz);
            float my = fmaxf(cy, 31.f - cy);
            float mx = fmaxf(cx, 31.f - cx);
            float maxd = sqrtf(mz * mz + my * my + mx * mx);
            cmeta[b * NE + r] = make_float4(cz, cy, cx, 1.f / (maxd + 1e-8f));
        }
        __syncthreads();
    }
}

// ---------------------------------------------------------------------------
// Kernel 3: WMMA GEMM with double-buffered LDS.
//   B tile (dense 128x32 f16, row stride 384) : TDM async DMA (wave 0 issues)
//   A tile (gathered center rows)             : register-pipelined coop stage
// One __syncthreads per K-step; DMA + A loads overlap WMMA of current step.
// ---------------------------------------------------------------------------
__global__ void __launch_bounds__(256)
k_gemm(const _Float16* __restrict__ nodes_h,
       const int* __restrict__ cidx,
       const float4* __restrict__ cmeta,
       const float* __restrict__ temp,
       float* __restrict__ combined)
{
    __shared__ _Float16 sA[2][NE * 32];    // centers tile [m][k]  2 x 8 KB
    __shared__ _Float16 sB[2][128 * 32];   // nodes   tile [n][k]  2 x 8 KB
    __shared__ int      s_ci[NE];
    __shared__ float4   s_md[NE];

    int t = threadIdx.x;
    int b    = blockIdx.y;
    int nblk = blockIdx.x;
    int w = t >> 5, lane = t & 31;
    int laneLo = lane & 15, laneHi = lane >> 4;

    if (t < NE) { s_ci[t] = cidx[b * NE + t]; s_md[t] = cmeta[b * NE + t]; }
    __syncthreads();

    int mrow = t >> 1;                    // 0..127 staging row
    int kp   = (t & 1) * 16;              // 0 or 16 halves
    long abase = (long)b * NN;
    const _Float16* srcA = nodes_h + (abase + s_ci[mrow]) * EMBED + kp;
    const _Float16* gB   = nodes_h + (abase + (long)nblk * 128) * EMBED;
#if !USE_TDM
    const _Float16* srcB = gB + (long)mrow * EMBED + kp;
#endif

    v8f acc[8];
#pragma unroll
    for (int i = 0; i < 8; ++i) acc[i] = v8f{};

    // ---- prologue: stage step 0 into buffer 0 ----
    uint4 ra0 = ((const uint4*)srcA)[0];
    uint4 ra1 = ((const uint4*)srcA)[1];
    { uint4* da = (uint4*)(&sA[0][mrow * 32 + kp]); da[0] = ra0; da[1] = ra1; }
#if USE_TDM
    if (w == 0) tdm_load_b(gB, &sB[0][0]);
#else
    uint4 rb0 = ((const uint4*)srcB)[0];
    uint4 rb1 = ((const uint4*)srcB)[1];
    { uint4* db = (uint4*)(&sB[0][mrow * 32 + kp]); db[0] = rb0; db[1] = rb1; }
#endif

    const int NSTEP = EMBED / 32;         // 12
    for (int s = 0; s < NSTEP; ++s) {
        int cur = s & 1, nxt = cur ^ 1;
        int kk  = s * 32;
#if USE_TDM
        __builtin_amdgcn_s_wait_tensorcnt(0);   // current B tile landed in LDS
#endif
        __syncthreads();                        // publishes A stores + B tile

        bool more = (s + 1 < NSTEP);
        if (more) {
#if USE_TDM
            if (w == 0) tdm_load_b(gB + kk + 32, &sB[nxt][0]);
#else
            rb0 = ((const uint4*)(srcB + kk + 32))[0];
            rb1 = ((const uint4*)(srcB + kk + 32))[1];
#endif
            ra0 = ((const uint4*)(srcA + kk + 32))[0];
            ra1 = ((const uint4*)(srcA + kk + 32))[1];
            __builtin_prefetch(srcA + kk + 64, 0, 1);   // two steps ahead
        }

        // B fragment: lane column n = w*16 + laneLo; halves = K[16*laneHi..+15]
        V16H bf;
        const _Float16* bcol = &sB[cur][(w * 16 + laneLo) * 32 + 16 * laneHi];
        bf.h[0] = *(const v8h*)(bcol);
        bf.h[1] = *(const v8h*)(bcol + 8);

        // Preload all 8 A fragments, then issue 8 back-to-back WMMAs
        V16H af[8];
#pragma unroll
        for (int mt = 0; mt < 8; ++mt) {
            const _Float16* arow = &sA[cur][(mt * 16 + laneLo) * 32];
            af[mt].h[0] = *(const v8h*)(arow + 16 * laneHi);
            af[mt].h[1] = *(const v8h*)(arow + 16 * laneHi + 8);
        }
#pragma unroll
        for (int mt = 0; mt < 8; ++mt) {
            acc[mt] = __builtin_amdgcn_wmma_f32_16x16x32_f16(
                false, af[mt].v, false, bf.v, (short)0, acc[mt], false, false);
        }

        if (more) {   // stage next A tile (safe: barrier above proved buf free)
            uint4* da = (uint4*)(&sA[nxt][mrow * 32 + kp]);
            da[0] = ra0; da[1] = ra1;
#if !USE_TDM
            uint4* db = (uint4*)(&sB[nxt][mrow * 32 + kp]);
            db[0] = rb0; db[1] = rb1;
#endif
        }
    }

    // Epilogue: combined = 0.9*sem/temp + 0.1*(1 - dist/maxdist)
    float invT = 1.f / temp[0];
    int n = nblk * 128 + w * 16 + laneLo;
    float z  = 2.f * (float)(n >> 10);
    float y  = (float)((n >> 5) & 31);
    float xx = (float)(n & 31);
#pragma unroll
    for (int mt = 0; mt < 8; ++mt) {
#pragma unroll
        for (int r = 0; r < 8; ++r) {
            int e = mt * 16 + laneHi * 8 + r;     // D layout: M = r + 8*laneHi
            float4 md = s_md[e];
            float dz = z - md.x, dy = y - md.y, dx = xx - md.z;
            float dist = sqrtf(dz * dz + dy * dy + dx * dx) * md.w;
            combined[((long)b * NE + e) * NN + n] =
                0.9f * acc[mt][r] * invT + 0.1f * (1.f - dist);
        }
    }
}

// ---------------------------------------------------------------------------
// Kernel 4: per-(b,e) top-64 over combined row -> scatter H[b, n, e] = 1
// ---------------------------------------------------------------------------
__global__ void __launch_bounds__(256)
k_topk_scatter(const float* __restrict__ combined,
               float* __restrict__ out)
{
    __shared__ float smax[256];
    __shared__ int   sidx[256];
    int t = threadIdx.x;
    int e = blockIdx.x;
    int b = blockIdx.y;
    const float* row = combined + ((long)b * NE + e) * NN;

    float vals[64];
#pragma unroll
    for (int j = 0; j < 64; ++j) vals[j] = row[j * 256 + t];

    for (int r = 0; r < TOPKK; ++r) {
        float best = -1e30f; int bj = 0;
#pragma unroll
        for (int j = 0; j < 64; ++j) if (vals[j] > best) { best = vals[j]; bj = j; }
        smax[t] = best; sidx[t] = bj * 256 + t;
        __syncthreads();
        for (int s = 128; s >= 1; s >>= 1) {
            if (t < s) {
                float om = smax[t + s]; int oi = sidx[t + s];
                if (om > smax[t] || (om == smax[t] && oi < sidx[t])) {
                    smax[t] = om; sidx[t] = oi;
                }
            }
            __syncthreads();
        }
        int nbest = sidx[0];
        if ((nbest & 255) == t) {
            int jj = nbest >> 8;
#pragma unroll
            for (int j = 0; j < 64; ++j) if (j == jj) vals[j] = -1e30f;
        }
        if (t == 0) out[((long)b * NN + nbest) * NE + e] = 1.0f;
        __syncthreads();
    }
}

// ---------------------------------------------------------------------------
extern "C" void kernel_launch(void* const* d_in, const int* in_sizes, int n_in,
                              void* d_out, int out_size, void* d_ws, size_t ws_size,
                              hipStream_t stream) {
    const float* nodes = (const float*)d_in[0];   // (B,N,C) f32
    const float* ct    = (const float*)d_in[1];   // (1,1,C) f32
    const float* temp  = (const float*)d_in[2];   // (1,)    f32
    float* out = (float*)d_out;                   // (B,N,Ne) f32

    // workspace partition (all 16B aligned)
    char* ws = (char*)d_ws;
    _Float16* nodes_h = (_Float16*)ws;                 // B*N*C f16   = 50,331,648 B
    size_t off = (size_t)NB * NN * EMBED * 2;
    float* scores = (float*)(ws + off); off += (size_t)NB * NN * 4;     // 262,144 B
    int*   cidx   = (int*)(ws + off);   off += (size_t)NB * NE * 4;     //   2,048 B
    float4* cmeta = (float4*)(ws + off); off += (size_t)NB * NE * 16;   //   8,192 B
    float* combined = (float*)(ws + off);              // B*Ne*N f32  = 33,554,432 B

    long n4 = (long)out_size >> 2;
    k_zero<<<dim3((unsigned)((n4 + 255) / 256)), 256, 0, stream>>>(out, n4);
    k_normalize<<<dim3(NB * NN / 8), 256, 0, stream>>>(nodes, ct, temp, nodes_h, scores);
    k_topk_centers<<<dim3(NB), 256, 0, stream>>>(scores, cidx, cmeta);
    k_gemm<<<dim3(NN / 128, NB), 256, 0, stream>>>(nodes_h, cidx, cmeta, temp, combined);
    k_topk_scatter<<<dim3(NE, NB), 256, 0, stream>>>(combined, out);
}
